// MultiHeadAttention_71262097375551
// MI455X (gfx1250) — compile-verified, matrix operand order checked by reading, not run on hardware
//
#include <hip/hip_runtime.h>

// ---------------------------------------------------------------------------
// MI455X (gfx1250) multi-head attention, bf16 WMMA pipeline.
// Shapes: B=4, S=2048, D=1024, H=16, dk=dv=64.
// GEMMs tiled 128x128 per block (8 WMMAs/wave/k-step); attention stages K via
// the Tensor Data Mover (TENSORcnt) and V^T via async-to-LDS (ASYNCcnt),
// double-buffered with one barrier per 32-key chunk.
// ---------------------------------------------------------------------------

typedef __bf16 bf16_t;
typedef __attribute__((ext_vector_type(16))) __bf16 v16bf;
typedef __attribute__((ext_vector_type(8)))  float  v8f;
typedef __attribute__((ext_vector_type(4)))  uint32_t v4u;
typedef __attribute__((ext_vector_type(8)))  uint32_t v8u;

constexpr int Bc   = 4;
constexpr int Sc   = 2048;
constexpr int Dc   = 1024;
constexpr int Hc   = 16;
constexpr int DKc  = 64;
constexpr int BHc  = Bc * Hc;      // 64
constexpr int MTOT = Bc * Sc;      // 8192

// --- CDNA5 async LDS copy helpers (VFLAT async ops, tracked by ASYNCcnt) ---
__device__ __forceinline__ void async_copy_b128(void* lds_dst, const void* gsrc) {
    // generic shared pointer low 32 bits == wave-relative LDS byte address
    uint32_t lds = (uint32_t)(size_t)lds_dst;
    uint64_t ga  = (uint64_t)(size_t)gsrc;
    asm volatile("global_load_async_to_lds_b128 %0, %1, off"
                 :: "v"(lds), "v"(ga) : "memory");
}
__device__ __forceinline__ void wait_asynccnt0() {
    asm volatile("s_wait_asynccnt 0x0" ::: "memory");
}

// --- CDNA5 Tensor Data Mover: 2D tile (rows x rowElems bf16) -> LDS ---
// D# built per ISA 8.3/8.4: group0 = {count, lds_addr, global_addr, type=2},
// group1 = {data_size=2B, tensor dims, tile dims, row stride}.  Issued by a
// single wave (EXEC ignored), tracked with TENSORcnt.
__device__ __forceinline__ void tdm_load_2d(void* lds_dst, const void* gsrc,
                                            uint32_t rowElems, uint32_t rows,
                                            uint32_t rowStrideElems) {
    uint32_t lds = (uint32_t)(size_t)lds_dst;
    uint64_t ga  = (uint64_t)(size_t)gsrc;
    v4u g0;
    g0[0] = 1u;                                   // count=1, user descriptor
    g0[1] = lds;                                  // lds_addr
    g0[2] = (uint32_t)ga;                         // global_addr[31:0]
    g0[3] = (uint32_t)(ga >> 32) | (2u << 30);    // global_addr[56:32] | type=2
    v8u g1;
    g1[0] = (1u << 16);                           // data_size=1 (2 bytes)
    g1[1] = (rowElems << 16);                     // tensor_dim0[15:0] @ bits63:48
    g1[2] = (rowElems >> 16) | (rows << 16);      // dim0[31:16] | tensor_dim1[15:0]
    g1[3] = (rowElems << 16);                     // dim1[31:16]=0 | tile_dim0
    g1[4] = rows;                                 // tile_dim1 | tile_dim2=0
    g1[5] = rowStrideElems;                       // tensor_dim0_stride[31:0]
    g1[6] = 0u;                                   // stride hi | dim1_stride lo
    g1[7] = 0u;                                   // dim1_stride hi
    asm volatile("tensor_load_to_lds %0, %1" :: "s"(g0), "s"(g1) : "memory");
}
__device__ __forceinline__ void wait_tensorcnt0() {
    asm volatile("s_wait_tensorcnt 0x0" ::: "memory");
}

// ---------------------------------------------------------------------------
// Kernel A: fused Q/K/V projection GEMMs (z = 0/1/2 selects matrix).
// C[8192,1024] = X[8192,1024] @ W[1024,1024] + b, emitted bf16.
// Q,K land head-major [B*H][S][64]; V lands TRANSPOSED [B*H][64][S] so the
// attention kernel can stage both K and V^T with pure row copies.
// Block = 256 thr (8 waves); tile 128(M) x 128(N); wave grid 2x4, 8 accs.
// ---------------------------------------------------------------------------
__global__ __launch_bounds__(256)
void qkv_proj_kernel(const float* __restrict__ Xq, const float* __restrict__ Xk,
                     const float* __restrict__ Xv,
                     const float* __restrict__ Wq, const float* __restrict__ bq,
                     const float* __restrict__ Wk, const float* __restrict__ bk,
                     const float* __restrict__ Wv, const float* __restrict__ bv,
                     bf16_t* __restrict__ Qw, bf16_t* __restrict__ Kw,
                     bf16_t* __restrict__ Vw)
{
    __shared__ __align__(32) bf16_t lA[128 * 32];   // [m][k]
    __shared__ __align__(32) bf16_t lB[128 * 32];   // [n][k] (transposed)

    const int z = blockIdx.z;
    const float* X   = (z == 0) ? Xq : (z == 1) ? Xk : Xv;
    const float* W   = (z == 0) ? Wq : (z == 1) ? Wk : Wv;
    const float* bia = (z == 0) ? bq : (z == 1) ? bk : bv;
    bf16_t*      dst = (z == 0) ? Qw : (z == 1) ? Kw : Vw;

    const int t    = threadIdx.x;
    const int wave = t >> 5;
    const int lane = t & 31;
    const int lo   = lane & 15;
    const int hi   = lane >> 4;
    const int wm   = wave >> 2;        // 0..1  (64 rows each)
    const int wn   = wave & 3;         // 0..3  (32 cols each)
    const int m0   = blockIdx.x * 128;
    const int n0   = blockIdx.y * 128;

    v8f acc[4][2];
    #pragma unroll
    for (int i = 0; i < 4; ++i)
        #pragma unroll
        for (int j = 0; j < 2; ++j) acc[i][j] = (v8f){};

    for (int kc = 0; kc < Dc; kc += 32) {
        // --- stage A tile: 128x32 fp32 -> bf16 (each thread: 16 contiguous) ---
        {
            const int row = t >> 1;
            const int cb  = (t & 1) * 16;
            const float* src = X + (size_t)(m0 + row) * Dc + kc + cb;
            __builtin_prefetch(src + 32, 0, 0);       // global_prefetch_b8
            #pragma unroll
            for (int j = 0; j < 16; j += 4) {
                float4 f = *(const float4*)(src + j);
                bf16_t* d4 = &lA[row * 32 + cb + j];
                d4[0] = (bf16_t)f.x; d4[1] = (bf16_t)f.y;
                d4[2] = (bf16_t)f.z; d4[3] = (bf16_t)f.w;
            }
        }
        // --- stage B tile transposed: lB[n][k] = W[kc+k][n0+n], 32x128 src ---
        {
            const int k  = t >> 3;                    // 0..31
            const int nb = (t & 7) * 16;              // 0..112
            const float* wrow = W + (size_t)(kc + k) * Dc + n0 + nb;
            __builtin_prefetch(wrow + 32 * Dc, 0, 0);
            #pragma unroll
            for (int jj = 0; jj < 4; ++jj) {
                float4 f = *(const float4*)(wrow + jj * 4);
                lB[(nb + jj * 4 + 0) * 32 + k] = (bf16_t)f.x;
                lB[(nb + jj * 4 + 1) * 32 + k] = (bf16_t)f.y;
                lB[(nb + jj * 4 + 2) * 32 + k] = (bf16_t)f.z;
                lB[(nb + jj * 4 + 3) * 32 + k] = (bf16_t)f.w;
            }
        }
        __syncthreads();

        v16bf am[4], bn[2];
        #pragma unroll
        for (int i = 0; i < 4; ++i)
            am[i] = *(const v16bf*)&lA[(wm * 64 + i * 16 + lo) * 32 + hi * 16];
        #pragma unroll
        for (int j = 0; j < 2; ++j)
            bn[j] = *(const v16bf*)&lB[(wn * 32 + j * 16 + lo) * 32 + hi * 16];

        #pragma unroll
        for (int i = 0; i < 4; ++i)
            #pragma unroll
            for (int j = 0; j < 2; ++j)
                acc[i][j] = __builtin_amdgcn_wmma_f32_16x16x32_bf16(
                    false, am[i], false, bn[j], (short)0, acc[i][j], false, false);
        __syncthreads();
    }

    // --- epilogue: bias + scatter.  Q/K: [B*H][S][64]; V^T: [B*H][64][S].
    // Layout folded into two uniform strides (no per-element selects):
    //   off = (bb*H + hh)*(S*64) + ss*sstr + dd*dstr
    const int bb     = m0 >> 11;              // blocks never straddle batches
    const int ssbase = (m0 & 2047) + wm * 64;
    const size_t sstr = (z == 2) ? 1 : DKc;
    const size_t dstr = (z == 2) ? Sc : 1;
    #pragma unroll
    for (int j = 0; j < 2; ++j) {
        const int col = n0 + wn * 32 + j * 16 + lo;
        const int hh  = col >> 6;
        const int dd  = col & 63;
        const float bval = bia[col];
        const size_t headoff = (size_t)(bb * Hc + hh) * (Sc * DKc) + (size_t)dd * dstr;
        #pragma unroll
        for (int i = 0; i < 4; ++i) {
            #pragma unroll
            for (int r = 0; r < 8; ++r) {
                const int ss = ssbase + i * 16 + r + (hi << 3);
                dst[headoff + (size_t)ss * sstr] = (bf16_t)(acc[i][j][r] + bval);
            }
        }
    }
}

// ---------------------------------------------------------------------------
// Kernel B: flash attention.  grid (S/128, B*H), 8 waves/block, each wave
// owns a 16-query tile; streams keys in chunks of 32 with online softmax.
// K chunk staged by the Tensor Data Mover (wave 0 issues one 2D descriptor,
// TENSORcnt); V^T chunk staged by GLOBAL_LOAD_ASYNC_TO_LDS_B128 (ASYNCcnt).
// Double-buffered; one barrier per chunk.
// ---------------------------------------------------------------------------
__global__ __launch_bounds__(256)
void attn_kernel(const bf16_t* __restrict__ Qw, const bf16_t* __restrict__ Kw,
                 const bf16_t* __restrict__ Vt,
                 const unsigned char* __restrict__ masks,
                 bf16_t* __restrict__ Ow)
{
    __shared__ __align__(32) bf16_t lK[2][32 * 64];   // [buf][key][d]
    __shared__ __align__(32) bf16_t lVt[2][64 * 32];  // [buf][d][key]
    __shared__ __align__(32) bf16_t lP[8][16 * 32];   // per-wave probs [row][key]

    const int t    = threadIdx.x;
    const int wave = t >> 5;
    const int lane = t & 31;
    const int lo   = lane & 15;
    const int hi   = lane >> 4;

    const int bh = blockIdx.y;
    const int bb = bh / Hc;
    const size_t base  = (size_t)bh * Sc * DKc;   // Q/K rows
    const size_t vbase = (size_t)bh * DKc * Sc;   // V^T rows
    const unsigned char* maskp = masks + (size_t)bb * Sc * Sc;

    const int q0 = blockIdx.x * 128 + wave * 16;

    // Q fragments for this wave's 16 queries (16x64 = two 16x32 A-fragments)
    const bf16_t* qrow = Qw + base + (size_t)(q0 + lo) * DKc;
    v16bf qa0 = *(const v16bf*)(qrow +  0 + hi * 16);
    v16bf qa1 = *(const v16bf*)(qrow + 32 + hi * 16);

    float mrun[8], lrun[8];
    #pragma unroll
    for (int r = 0; r < 8; ++r) { mrun[r] = -1e30f; lrun[r] = 0.f; }
    v8f o0 = {}, o1 = {}, o2 = {}, o3 = {};

    // V^T async staging: each thread one 16B copy per chunk
    const int dr  = t >> 2;             // 0..63 (V^T d row)
    const int dsz = (t & 3) * 8;        // elem offset within 32-elem Vt row

    auto issue_chunk = [&](int kc, int buf) {
        async_copy_b128(&lVt[buf][dr * 32 + dsz],
                        Vt + vbase + (size_t)dr * Sc + kc + dsz);
        if (wave == 0)   // TDM: one 32x64 bf16 tile, rows contiguous in LDS
            tdm_load_2d(&lK[buf][0], Kw + base + (size_t)kc * DKc,
                        /*rowElems=*/DKc, /*rows=*/32, /*rowStride=*/DKc);
    };

    issue_chunk(0, 0);

    for (int kc = 0; kc < Sc; kc += 32) {
        const int cur = (kc >> 5) & 1;
        wait_asynccnt0();               // this wave's V^T copies landed
        if (wave == 0) wait_tensorcnt0();   // K DMA landed
        __syncthreads();                // everyone's copies landed; prev reads done
        if (kc + 32 < Sc) issue_chunk(kc + 32, cur ^ 1);

        // --- scores: Q(16x64) @ K^T(64x32) -> two 16x16 tiles, 4 WMMAs ---
        v8f s0 = {}, s1 = {};
        {
            v16bf kb;
            kb = *(const v16bf*)&lK[cur][( 0 + lo) * 64 +  0 + hi * 16];
            s0 = __builtin_amdgcn_wmma_f32_16x16x32_bf16(false, qa0, false, kb, (short)0, s0, false, false);
            kb = *(const v16bf*)&lK[cur][( 0 + lo) * 64 + 32 + hi * 16];
            s0 = __builtin_amdgcn_wmma_f32_16x16x32_bf16(false, qa1, false, kb, (short)0, s0, false, false);
            kb = *(const v16bf*)&lK[cur][(16 + lo) * 64 +  0 + hi * 16];
            s1 = __builtin_amdgcn_wmma_f32_16x16x32_bf16(false, qa0, false, kb, (short)0, s1, false, false);
            kb = *(const v16bf*)&lK[cur][(16 + lo) * 64 + 32 + hi * 16];
            s1 = __builtin_amdgcn_wmma_f32_16x16x32_bf16(false, qa1, false, kb, (short)0, s1, false, false);
        }

        // --- online softmax (C-layout: row = r + 8*hi, col = lo) ---
        #pragma unroll
        for (int r = 0; r < 8; ++r) {
            const int q = q0 + r + 8 * hi;
            float v0 = s0[r] * 0.125f;                 // 1/sqrt(64)
            float v1 = s1[r] * 0.125f;
            if (!maskp[(size_t)q * Sc + kc +      lo]) v0 = -1e9f;
            if (!maskp[(size_t)q * Sc + kc + 16 + lo]) v1 = -1e9f;
            float rm = fmaxf(v0, v1);
            rm = fmaxf(rm, __shfl_xor(rm, 1, 32));
            rm = fmaxf(rm, __shfl_xor(rm, 2, 32));
            rm = fmaxf(rm, __shfl_xor(rm, 4, 32));
            rm = fmaxf(rm, __shfl_xor(rm, 8, 32));
            float mnew = fmaxf(mrun[r], rm);
            float corr = __expf(mrun[r] - mnew);
            float p0   = __expf(v0 - mnew);
            float p1   = __expf(v1 - mnew);
            mrun[r] = mnew;
            lrun[r] = lrun[r] * corr + p0 + p1;        // per-lane partial row sum
            o0[r] *= corr; o1[r] *= corr; o2[r] *= corr; o3[r] *= corr;
            lP[wave][(r + 8 * hi) * 32 +      lo] = (bf16_t)p0;
            lP[wave][(r + 8 * hi) * 32 + 16 + lo] = (bf16_t)p1;
        }

        // --- P(16x32) @ V(32x64): re-fragment P via LDS, 4 WMMAs ---
        v16bf pa = *(const v16bf*)&lP[wave][lo * 32 + hi * 16];
        v16bf vb;
        vb = *(const v16bf*)&lVt[cur][( 0 + lo) * 32 + hi * 16];
        o0 = __builtin_amdgcn_wmma_f32_16x16x32_bf16(false, pa, false, vb, (short)0, o0, false, false);
        vb = *(const v16bf*)&lVt[cur][(16 + lo) * 32 + hi * 16];
        o1 = __builtin_amdgcn_wmma_f32_16x16x32_bf16(false, pa, false, vb, (short)0, o1, false, false);
        vb = *(const v16bf*)&lVt[cur][(32 + lo) * 32 + hi * 16];
        o2 = __builtin_amdgcn_wmma_f32_16x16x32_bf16(false, pa, false, vb, (short)0, o2, false, false);
        vb = *(const v16bf*)&lVt[cur][(48 + lo) * 32 + hi * 16];
        o3 = __builtin_amdgcn_wmma_f32_16x16x32_bf16(false, pa, false, vb, (short)0, o3, false, false);
    }

    // --- finalize: reduce row sums across 16-lane halves, normalize, store ---
    #pragma unroll
    for (int r = 0; r < 8; ++r) {
        float ls = lrun[r];
        ls += __shfl_xor(ls, 1, 32);
        ls += __shfl_xor(ls, 2, 32);
        ls += __shfl_xor(ls, 4, 32);
        ls += __shfl_xor(ls, 8, 32);
        const float inv = 1.0f / ls;
        const int q = q0 + r + 8 * hi;
        bf16_t* orow = Ow + base + (size_t)q * DKc;
        orow[ 0 + lo] = (bf16_t)(o0[r] * inv);
        orow[16 + lo] = (bf16_t)(o1[r] * inv);
        orow[32 + lo] = (bf16_t)(o2[r] * inv);
        orow[48 + lo] = (bf16_t)(o3[r] * inv);
    }
}

// ---------------------------------------------------------------------------
// Kernel C: output projection with the reference's permutation quirk folded
// into the A-tile gather:  cat2[b,s,v] = headcat[b, (s*1024+v)%2048,
// (s*1024+v)/2048], where headcat[b,s,v] = Ow[b, v/64, s, v%64].
// out = cat2 @ Wo + bo  (fp32 output).  Tile 128x128, 8 WMMAs/wave/step.
// ---------------------------------------------------------------------------
__global__ __launch_bounds__(256)
void out_proj_kernel(const bf16_t* __restrict__ Ow, const float* __restrict__ Wo,
                     const float* __restrict__ bo, float* __restrict__ out)
{
    __shared__ __align__(32) bf16_t lA[128 * 32];
    __shared__ __align__(32) bf16_t lB[128 * 32];

    const int t    = threadIdx.x;
    const int wave = t >> 5;
    const int lane = t & 31;
    const int lo   = lane & 15;
    const int hi   = lane >> 4;
    const int wm   = wave >> 2;
    const int wn   = wave & 3;
    const int m0   = blockIdx.x * 128;
    const int n0   = blockIdx.y * 128;

    v8f acc[4][2];
    #pragma unroll
    for (int i = 0; i < 4; ++i)
        #pragma unroll
        for (int j = 0; j < 2; ++j) acc[i][j] = (v8f){};

    for (int kc = 0; kc < Dc; kc += 32) {
        // --- gather A tile through the permutation ---
        #pragma unroll
        for (int i = 0; i < 16; ++i) {
            int e   = t * 16 + i;           // 0..4095
            int row = e >> 5;
            int c   = e & 31;
            int m   = m0 + row;
            int bb  = m >> 11;
            int ss  = m & 2047;
            int idx = ss * 1024 + kc + c;
            int s2  = idx & 2047;
            int v2  = idx >> 11;
            int h2  = v2 >> 6;
            int d2  = v2 & 63;
            lA[row * 32 + c] = Ow[(size_t)((bb * Hc + h2) * Sc + s2) * DKc + d2];
        }
        // --- stage B tile transposed: lB[n][k] = Wo[kc+k][n0+n] ---
        {
            const int k  = t >> 3;
            const int nb = (t & 7) * 16;
            const float* wrow = Wo + (size_t)(kc + k) * Dc + n0 + nb;
            __builtin_prefetch(wrow + 32 * Dc, 0, 0);
            #pragma unroll
            for (int jj = 0; jj < 4; ++jj) {
                float4 f = *(const float4*)(wrow + jj * 4);
                lB[(nb + jj * 4 + 0) * 32 + k] = (bf16_t)f.x;
                lB[(nb + jj * 4 + 1) * 32 + k] = (bf16_t)f.y;
                lB[(nb + jj * 4 + 2) * 32 + k] = (bf16_t)f.z;
                lB[(nb + jj * 4 + 3) * 32 + k] = (bf16_t)f.w;
            }
        }
        __syncthreads();

        v16bf am[4], bn[2];
        #pragma unroll
        for (int i = 0; i < 4; ++i)
            am[i] = *(const v16bf*)&lA[(wm * 64 + i * 16 + lo) * 32 + hi * 16];
        #pragma unroll
        for (int j = 0; j < 2; ++j)
            bn[j] = *(const v16bf*)&lB[(wn * 32 + j * 16 + lo) * 32 + hi * 16];

        #pragma unroll
        for (int i = 0; i < 4; ++i)
            #pragma unroll
            for (int j = 0; j < 2; ++j)
                acc[i][j] = __builtin_amdgcn_wmma_f32_16x16x32_bf16(
                    false, am[i], false, bn[j], (short)0, acc[i][j], false, false);
        __syncthreads();
    }

    #pragma unroll
    for (int j = 0; j < 2; ++j) {
        const int col  = n0 + wn * 32 + j * 16 + lo;
        const float bv = bo[col];
        #pragma unroll
        for (int i = 0; i < 4; ++i) {
            #pragma unroll
            for (int r = 0; r < 8; ++r) {
                int mg = m0 + wm * 64 + i * 16 + r + (hi << 3);
                out[(size_t)mg * Dc + col] = acc[i][j][r] + bv;
            }
        }
    }
}

// ---------------------------------------------------------------------------
// Host launcher
// ---------------------------------------------------------------------------
extern "C" void kernel_launch(void* const* d_in, const int* in_sizes, int n_in,
                              void* d_out, int out_size, void* d_ws, size_t ws_size,
                              hipStream_t stream) {
    const float* queries = (const float*)d_in[0];
    const float* keys    = (const float*)d_in[1];
    const float* values  = (const float*)d_in[2];
    const unsigned char* masks = (const unsigned char*)d_in[3];
    const float* Wq = (const float*)d_in[4];
    const float* bq = (const float*)d_in[5];
    const float* Wk = (const float*)d_in[6];
    const float* bk = (const float*)d_in[7];
    const float* Wv = (const float*)d_in[8];
    const float* bv = (const float*)d_in[9];
    const float* Wo = (const float*)d_in[10];
    const float* bo = (const float*)d_in[11];
    float* out = (float*)d_out;

    // workspace: 4 x (B*H*S*64) bf16 = 4 x 16 MB
    const size_t tensElems = (size_t)Bc * Hc * Sc * DKc;   // 8,388,608
    bf16_t* Qw = (bf16_t*)d_ws;
    bf16_t* Kw = Qw + tensElems;
    bf16_t* Vw = Kw + tensElems;      // stored transposed [B*H][64][S]
    bf16_t* Ow = Vw + tensElems;

    dim3 blk(256);
    dim3 gA(MTOT / 128, Dc / 128, 3);
    qkv_proj_kernel<<<gA, blk, 0, stream>>>(queries, keys, values,
                                            Wq, bq, Wk, bk, Wv, bv,
                                            Qw, Kw, Vw);

    dim3 gB(Sc / 128, BHc);
    attn_kernel<<<gB, blk, 0, stream>>>(Qw, Kw, Vw, masks, Ow);

    dim3 gC(MTOT / 128, Dc / 128);
    out_proj_kernel<<<gC, blk, 0, stream>>>(Ow, Wo, bo, out);
}
